// ImageCaptionModel_68702296866975
// MI455X (gfx1250) — matrix operand order repaired and checked by reading, not compiled, not run on hardware
//
#include <hip/hip_runtime.h>
#include <hip/hip_bf16.h>

// Problem dims (fixed by reference)
#define B_   64
#define T_   48
#define V_   10000
#define E_   512
#define H_   1024
#define L_   512
#define C_   2048
#define NNM  512        // conv out channels
#define FAN  2048       // E + NNM + H
#define G4H  4096       // 4*H

typedef __attribute__((ext_vector_type(16))) __bf16 bf16x16;
typedef __attribute__((ext_vector_type(8)))  float  f32x8;

// Native converts -> v_cvt_pk_bf16_f32 on gfx1250.
__device__ __forceinline__ unsigned short f2bf(float f) {
  __bf16 h = (__bf16)f;
  return __builtin_bit_cast(unsigned short, h);
}
__device__ __forceinline__ float bf2f(unsigned short s) {
  return (float)__builtin_bit_cast(__bf16, s);
}

union Frag32 { int4 q[2]; bf16x16 v; __bf16 b[16]; };  // 32B == one wmma A/B fragment/lane

// B fragment (32x16 bf16, repacked K-contiguous): 16 contiguous halves per lane.
__device__ __forceinline__ bf16x16 load_b_frag(const unsigned short* p) {
  Frag32 u;
  u.q[0] = *reinterpret_cast<const int4*>(p);
  u.q[1] = *reinterpret_cast<const int4*>(p + 8);
  return u.v;
}
// A fragment (16x32 bf16) from a K-contiguous bf16 row: two 8-half runs at
// K = kA0.. and K = kA0+8..  (kA0 = 0 for lanes 0-15, 16 for lanes 16-31).
__device__ __forceinline__ bf16x16 load_a_frag_bf16(const unsigned short* row, int kA0) {
  Frag32 u;
  u.q[0] = *reinterpret_cast<const int4*>(row + kA0);
  u.q[1] = *reinterpret_cast<const int4*>(row + kA0 + 8);
  return u.v;
}
// Same A fragment from an f32 row; native casts -> packed bf16 cvt instructions.
__device__ __forceinline__ bf16x16 load_a_frag_f32(const float* row, int kA0) {
  float4 a0 = *reinterpret_cast<const float4*>(row + kA0);
  float4 a1 = *reinterpret_cast<const float4*>(row + kA0 + 4);
  float4 a2 = *reinterpret_cast<const float4*>(row + kA0 + 8);
  float4 a3 = *reinterpret_cast<const float4*>(row + kA0 + 12);
  Frag32 u;
  u.b[0]=(__bf16)a0.x;  u.b[1]=(__bf16)a0.y;  u.b[2]=(__bf16)a0.z;  u.b[3]=(__bf16)a0.w;
  u.b[4]=(__bf16)a1.x;  u.b[5]=(__bf16)a1.y;  u.b[6]=(__bf16)a1.z;  u.b[7]=(__bf16)a1.w;
  u.b[8]=(__bf16)a2.x;  u.b[9]=(__bf16)a2.y;  u.b[10]=(__bf16)a2.z; u.b[11]=(__bf16)a2.w;
  u.b[12]=(__bf16)a3.x; u.b[13]=(__bf16)a3.y; u.b[14]=(__bf16)a3.z; u.b[15]=(__bf16)a3.w;
  return u.v;
}
__device__ __forceinline__ f32x8 wmma_bf16(bf16x16 a, bf16x16 b, f32x8 c) {
  return __builtin_amdgcn_wmma_f32_16x16x32_bf16(false, a, false, b, (short)0, c, false, false);
}

// ---------------- prep kernels ----------------

__global__ void k_cvt_bf16(const float* __restrict__ src, unsigned short* __restrict__ dst, int n) {
  int i = blockIdx.x * blockDim.x + threadIdx.x;
  if (i < n) dst[i] = f2bf(src[i]);
}

// Repack row-major [K][N] f32 weights into WMMA B-fragment order, bf16:
// dst[(((ntile*ksteps)+kstep)*32 + lane)*16 + h] = W[kstep*32 + (lane>=16?16:0)+h][ntile*16 + lane%16]
__global__ void k_pack_B(const float* __restrict__ src, unsigned short* __restrict__ dst,
                         int N, int ksteps, long total) {
  long i = (long)blockIdx.x * blockDim.x + threadIdx.x;
  if (i >= total) return;
  int h    = (int)(i & 15);
  int lane = (int)((i >> 4) & 31);
  long rest = i >> 9;
  int kstep = (int)(rest % ksteps);
  long ntile = rest / ksteps;
  int K = kstep * 32 + ((lane >> 4) << 4) + h;
  long n = ntile * 16 + (lane & 15);
  dst[i] = f2bf(src[(size_t)K * N + n]);
}

// ---------------- conv (1x1) GEMM: Y[bl][o] = X[bl][:] . conv_w[o][:] ----------------
// M = B*L = 32768 (2048 mtiles), N = 512 (8 groups of 4 ntiles), K = 2048 (64 ksteps).
// Software-pipelined; launch_bounds(.,1) -> full VGPR budget, no spills.
__global__ void __launch_bounds__(256, 1)
k_conv_gemm(const float* __restrict__ X,
            const unsigned short* __restrict__ Wb,
            unsigned short* __restrict__ Ybf) {
  int lane = threadIdx.x & 31;
  int wave = blockIdx.x * (blockDim.x >> 5) + (threadIdx.x >> 5);
  int mtile  = wave >> 3;                  // 2048
  int ngroup = wave & 7;                   // 8 groups of 4 ntiles
  const float* arow = X + (size_t)(mtile * 16 + (lane & 15)) * C_;
  int kA0 = (lane >= 16) ? 16 : 0;
  int kB0 = (lane >= 16) ? 16 : 0;
  const unsigned short *br0, *br1, *br2, *br3;
  br0 = Wb + (size_t)((ngroup * 4 + 0) * 16 + (lane & 15)) * C_ + kB0;
  br1 = Wb + (size_t)((ngroup * 4 + 1) * 16 + (lane & 15)) * C_ + kB0;
  br2 = Wb + (size_t)((ngroup * 4 + 2) * 16 + (lane & 15)) * C_ + kB0;
  br3 = Wb + (size_t)((ngroup * 4 + 3) * 16 + (lane & 15)) * C_ + kB0;
  f32x8 acc0 = {0.f,0.f,0.f,0.f,0.f,0.f,0.f,0.f};
  f32x8 acc1 = acc0, acc2 = acc0, acc3 = acc0;
  // prologue
  bf16x16 a  = load_a_frag_f32(arow, kA0);
  bf16x16 b0 = load_b_frag(br0), b1 = load_b_frag(br1),
          b2 = load_b_frag(br2), b3 = load_b_frag(br3);
  for (int ks = 0; ks < C_ / 32 - 1; ++ks) {
    int kb = (ks + 1) * 32;
    bf16x16 an  = load_a_frag_f32(arow + kb, kA0);   // next-iteration loads first
    bf16x16 bn0 = load_b_frag(br0 + kb);
    bf16x16 bn1 = load_b_frag(br1 + kb);
    bf16x16 bn2 = load_b_frag(br2 + kb);
    bf16x16 bn3 = load_b_frag(br3 + kb);
    acc0 = wmma_bf16(a, b0, acc0);
    acc1 = wmma_bf16(a, b1, acc1);
    acc2 = wmma_bf16(a, b2, acc2);
    acc3 = wmma_bf16(a, b3, acc3);
    a = an; b0 = bn0; b1 = bn1; b2 = bn2; b3 = bn3;
  }
  acc0 = wmma_bf16(a, b0, acc0);
  acc1 = wmma_bf16(a, b1, acc1);
  acc2 = wmma_bf16(a, b2, acc2);
  acc3 = wmma_bf16(a, b3, acc3);
  int rbase = mtile * 16 + ((lane >= 16) ? 8 : 0);
  int col0 = ngroup * 64 + (lane & 15);
#pragma unroll
  for (int r = 0; r < 8; ++r) {
    size_t row = (size_t)(rbase + r) * NNM;
    Ybf[row + col0]      = f2bf(acc0[r]);
    Ybf[row + col0 + 16] = f2bf(acc1[r]);
    Ybf[row + col0 + 32] = f2bf(acc2[r]);
    Ybf[row + col0 + 48] = f2bf(acc3[r]);
  }
}

// ---------------- BN stats / scale / pcf ----------------

__global__ void k_bn_stats(const unsigned short* __restrict__ Ybf,
                           float* __restrict__ sum, float* __restrict__ sq) {
  int o = blockIdx.x;                       // 512 channels
  float s = 0.f, s2 = 0.f;
  for (int i = threadIdx.x; i < B_ * L_; i += blockDim.x) {
    float v = bf2f(Ybf[(size_t)i * NNM + o]);
    s += v; s2 += v * v;
  }
  __shared__ float sh[256], sh2[256];
  sh[threadIdx.x] = s; sh2[threadIdx.x] = s2; __syncthreads();
  for (int off = 128; off; off >>= 1) {
    if ((int)threadIdx.x < off) { sh[threadIdx.x] += sh[threadIdx.x + off];
                                  sh2[threadIdx.x] += sh2[threadIdx.x + off]; }
    __syncthreads();
  }
  if (!threadIdx.x) { sum[o] = sh[0]; sq[o] = sh2[0]; }
}

__global__ void k_bn_scale(const float* __restrict__ sum, const float* __restrict__ sq,
                           const float* __restrict__ gamma, const float* __restrict__ beta,
                           float* __restrict__ scale, float* __restrict__ shift) {
  int o = blockIdx.x * blockDim.x + threadIdx.x;
  if (o >= NNM) return;
  float inv = 1.0f / (float)(B_ * L_);
  float mu  = sum[o] * inv;
  float var = sq[o] * inv - mu * mu;
  float sc  = gamma[o] * rsqrtf(var + 1e-5f);
  scale[o] = sc;
  shift[o] = beta[o] - mu * sc;             // conv_b cancels exactly inside BN
}

__global__ void k_pcf(const unsigned short* __restrict__ Ybf,
                      const float* __restrict__ scale, const float* __restrict__ shift,
                      unsigned short* __restrict__ pcf) {
  int bl = blockIdx.x;                      // 32768 (b*L + l)
  const unsigned short* row = Ybf + (size_t)bl * NNM;
  float m = -3.4e38f;
  for (int o = threadIdx.x; o < NNM; o += blockDim.x) {
    float v = scale[o] * bf2f(row[o]) + shift[o];
    v = v > 0.f ? v : 0.01f * v;            // LeakyReLU
    m = fmaxf(m, v);
  }
  __shared__ float sh[256];
  sh[threadIdx.x] = m; __syncthreads();
  for (int off = 128; off; off >>= 1) {
    if ((int)threadIdx.x < off) sh[threadIdx.x] = fmaxf(sh[threadIdx.x], sh[threadIdx.x + off]);
    __syncthreads();
  }
  if (!threadIdx.x) pcf[bl] = f2bf(sh[0]);
}

// ---------------- LSTM ----------------

__global__ void k_init_state(unsigned short* __restrict__ h_bf, float* __restrict__ cbuf) {
  int i = blockIdx.x * blockDim.x + threadIdx.x;   // B_*H_ = 65536
  h_bf[i] = 0; cbuf[i] = 0.f;
}

// gates GEMM: G = [emb|pcf|h] @ W_gates   (M=64, N=4096, K=2048)
// waves: 4 mtiles x 64 ngroups (4 ntiles each); pipelined like conv.
__global__ void __launch_bounds__(256, 1)
k_gates_gemm(const unsigned short* __restrict__ emb_bf,
             const unsigned short* __restrict__ pcf_bf,
             const unsigned short* __restrict__ h_bf,
             const unsigned short* __restrict__ wg_pack,
             const int* __restrict__ tokens, int t,
             float* __restrict__ g) {
  int lane = threadIdx.x & 31;
  int wave = blockIdx.x * (blockDim.x >> 5) + (threadIdx.x >> 5);  // 0..255
  int mtile  = wave >> 6;                   // 4 mtiles
  int ngroup = wave & 63;                   // 64 groups of 4 ntiles
  int brow = mtile * 16 + (lane & 15);      // batch row this lane supplies A for
  int tok = tokens[brow * T_ + t];
  const unsigned short* eseg = emb_bf + (size_t)tok  * E_;
  const unsigned short* pseg = pcf_bf + (size_t)brow * NNM;
  const unsigned short* hseg = h_bf  + (size_t)brow * H_;
  int kA0 = (lane >= 16) ? 16 : 0;
  // B pointers: 4 packed ntile streams, 512 halves per kstep
  const unsigned short* bp0 = wg_pack + ((size_t)((ngroup * 4 + 0) * 64) * 32 + lane) * 16;
  const unsigned short* bp1 = wg_pack + ((size_t)((ngroup * 4 + 1) * 64) * 32 + lane) * 16;
  const unsigned short* bp2 = wg_pack + ((size_t)((ngroup * 4 + 2) * 64) * 32 + lane) * 16;
  const unsigned short* bp3 = wg_pack + ((size_t)((ngroup * 4 + 3) * 64) * 32 + lane) * 16;
  auto aseg = [&](int kb) -> const unsigned short* {
    if (kb < 512)  return eseg + kb;                 // embedding (K 0..511)
    if (kb < 1024) return pseg + (kb - 512);         // pcf       (K 512..1023)
    return hseg + (kb - 1024);                       // hidden    (K 1024..2047)
  };
  f32x8 acc0 = {0.f,0.f,0.f,0.f,0.f,0.f,0.f,0.f};
  f32x8 acc1 = acc0, acc2 = acc0, acc3 = acc0;
  bf16x16 a  = load_a_frag_bf16(aseg(0), kA0);
  bf16x16 b0 = load_b_frag(bp0), b1 = load_b_frag(bp1),
          b2 = load_b_frag(bp2), b3 = load_b_frag(bp3);
  for (int ks = 0; ks < 63; ++ks) {
    int kb = (ks + 1) * 32;
    size_t bo = (size_t)(ks + 1) * 512;
    bf16x16 an  = load_a_frag_bf16(aseg(kb), kA0);
    bf16x16 bn0 = load_b_frag(bp0 + bo);
    bf16x16 bn1 = load_b_frag(bp1 + bo);
    bf16x16 bn2 = load_b_frag(bp2 + bo);
    bf16x16 bn3 = load_b_frag(bp3 + bo);
    acc0 = wmma_bf16(a, b0, acc0);
    acc1 = wmma_bf16(a, b1, acc1);
    acc2 = wmma_bf16(a, b2, acc2);
    acc3 = wmma_bf16(a, b3, acc3);
    a = an; b0 = bn0; b1 = bn1; b2 = bn2; b3 = bn3;
  }
  acc0 = wmma_bf16(a, b0, acc0);
  acc1 = wmma_bf16(a, b1, acc1);
  acc2 = wmma_bf16(a, b2, acc2);
  acc3 = wmma_bf16(a, b3, acc3);
  int rbase = mtile * 16 + ((lane >= 16) ? 8 : 0);
  int col0 = ngroup * 64 + (lane & 15);
#pragma unroll
  for (int r = 0; r < 8; ++r) {
    size_t row = (size_t)(rbase + r) * G4H;
    g[row + col0]      = acc0[r];
    g[row + col0 + 16] = acc1[r];
    g[row + col0 + 32] = acc2[r];
    g[row + col0 + 48] = acc3[r];
  }
}

__global__ void k_lstm_cell(const float* __restrict__ g, const float* __restrict__ bg,
                            float* __restrict__ cbuf, float* __restrict__ hbuf,
                            unsigned short* __restrict__ h_bf) {
  int i = blockIdx.x * blockDim.x + threadIdx.x;   // 65536
  int b = i >> 10, nh = i & 1023;
  const float* gr = g + (size_t)b * G4H;
  float gi = gr[nh]          + bg[nh];
  float go = gr[H_ + nh]     + bg[H_ + nh];
  float gf = gr[2 * H_ + nh] + bg[2 * H_ + nh];
  float gc = gr[3 * H_ + nh] + bg[3 * H_ + nh];
  float it = 1.f / (1.f + __expf(-gi));
  float ot = 1.f / (1.f + __expf(-go));
  float ft = 1.f / (1.f + __expf(-gf));
  float ch = tanhf(gc);
  float c  = ft * cbuf[i] + it * ch;
  float h  = ot * tanhf(c);
  cbuf[i] = c; hbuf[i] = h; h_bf[i] = f2bf(h);
}

// logits GEMM: out[:,t,:] = h @ out_W + out_b   (M=64, N=10000=625*16, K=1024)
// waves: 4 mtiles x 157 ngroups (4 ntiles, tail clamped); pipelined.
__global__ void __launch_bounds__(256, 1)
k_logits_gemm(const unsigned short* __restrict__ h_bf,
              const unsigned short* __restrict__ wo_pack,
              const float* __restrict__ out_b,
              float* __restrict__ out, int t) {
  int lane = threadIdx.x & 31;
  int wave = blockIdx.x * (blockDim.x >> 5) + (threadIdx.x >> 5);
  if (wave >= 4 * 157) return;
  int mtile  = wave / 157;
  int ngroup = wave % 157;
  int nt0 = ngroup * 4;
  int nvalid = (625 - nt0) < 4 ? (625 - nt0) : 4;
  const unsigned short* arow = h_bf + (size_t)(mtile * 16 + (lane & 15)) * H_;
  int kA0 = (lane >= 16) ? 16 : 0;
  int n0 = nt0, n1 = nt0 + (nvalid > 1 ? 1 : 0),
      n2 = nt0 + (nvalid > 2 ? 2 : 0), n3 = nt0 + (nvalid > 3 ? 3 : 0);
  const unsigned short* bp0 = wo_pack + ((size_t)(n0 * 32) * 32 + lane) * 16;
  const unsigned short* bp1 = wo_pack + ((size_t)(n1 * 32) * 32 + lane) * 16;
  const unsigned short* bp2 = wo_pack + ((size_t)(n2 * 32) * 32 + lane) * 16;
  const unsigned short* bp3 = wo_pack + ((size_t)(n3 * 32) * 32 + lane) * 16;
  f32x8 acc0 = {0.f,0.f,0.f,0.f,0.f,0.f,0.f,0.f};
  f32x8 acc1 = acc0, acc2 = acc0, acc3 = acc0;
  bf16x16 a  = load_a_frag_bf16(arow, kA0);
  bf16x16 b0 = load_b_frag(bp0), b1 = load_b_frag(bp1),
          b2 = load_b_frag(bp2), b3 = load_b_frag(bp3);
  for (int ks = 0; ks < 31; ++ks) {
    size_t bo = (size_t)(ks + 1) * 512;
    bf16x16 an  = load_a_frag_bf16(arow + (ks + 1) * 32, kA0);
    bf16x16 bn0 = load_b_frag(bp0 + bo);
    bf16x16 bn1 = load_b_frag(bp1 + bo);
    bf16x16 bn2 = load_b_frag(bp2 + bo);
    bf16x16 bn3 = load_b_frag(bp3 + bo);
    acc0 = wmma_bf16(a, b0, acc0);
    acc1 = wmma_bf16(a, b1, acc1);
    acc2 = wmma_bf16(a, b2, acc2);
    acc3 = wmma_bf16(a, b3, acc3);
    a = an; b0 = bn0; b1 = bn1; b2 = bn2; b3 = bn3;
  }
  acc0 = wmma_bf16(a, b0, acc0);
  acc1 = wmma_bf16(a, b1, acc1);
  acc2 = wmma_bf16(a, b2, acc2);
  acc3 = wmma_bf16(a, b3, acc3);
  int rbase = mtile * 16 + ((lane >= 16) ? 8 : 0);
  const f32x8* accs[4] = {&acc0, &acc1, &acc2, &acc3};
#pragma unroll
  for (int j = 0; j < 4; ++j) {
    if (j >= nvalid) break;
    int col = (nt0 + j) * 16 + (lane & 15);
    float bias = out_b[col];
    const f32x8& av = *accs[j];
#pragma unroll
    for (int r = 0; r < 8; ++r)
      out[((size_t)(rbase + r) * T_ + t) * V_ + col] = av[r] + bias;
  }
}

__global__ void k_copy_h(const float* __restrict__ hbuf, float* __restrict__ out) {
  int i = blockIdx.x * blockDim.x + threadIdx.x;   // 65536
  out[(size_t)B_ * T_ * V_ + i] = hbuf[i];
}

// ---------------- launch ----------------

extern "C" void kernel_launch(void* const* d_in, const int* in_sizes, int n_in,
                              void* d_out, int out_size, void* d_ws, size_t ws_size,
                              hipStream_t stream) {
  const float* cnn   = (const float*)d_in[0];   // [B,L,C]
  const float* convw = (const float*)d_in[1];   // [512,2048]
  // d_in[2] conv_b: cancels inside BatchNorm (constant per-channel shift)
  const float* gamma = (const float*)d_in[3];
  const float* beta  = (const float*)d_in[4];
  const float* emb   = (const float*)d_in[5];   // [V,E]
  const float* Wg    = (const float*)d_in[6];   // [2048,4096]
  const float* bg    = (const float*)d_in[7];   // [4096]
  const float* Wo    = (const float*)d_in[8];   // [1024,10000]
  const float* bo    = (const float*)d_in[9];   // [10000]
  const int*   toks  = (const int*)d_in[10];    // [B,T]
  float* out = (float*)d_out;

  // workspace carve-out (256B aligned regions)
  char* p = (char*)d_ws;
  auto carve = [&](size_t bytes) { void* r = (void*)p; p += (bytes + 255) & ~(size_t)255; return r; };
  unsigned short* y_bf    = (unsigned short*)carve((size_t)B_ * L_ * NNM * 2);   // 32 MB
  unsigned short* wg_pack = (unsigned short*)carve((size_t)FAN * G4H * 2);       // 16 MB
  unsigned short* wo_pack = (unsigned short*)carve((size_t)H_ * V_ * 2);         // 20 MB
  unsigned short* emb_bf  = (unsigned short*)carve((size_t)V_ * E_ * 2);         // 10 MB
  unsigned short* cw_bf   = (unsigned short*)carve((size_t)NNM * C_ * 2);        // 2 MB
  float* chsum   = (float*)carve(NNM * 4);
  float* chsq    = (float*)carve(NNM * 4);
  float* scale   = (float*)carve(NNM * 4);
  float* shift   = (float*)carve(NNM * 4);
  unsigned short* pcf_bf = (unsigned short*)carve((size_t)B_ * L_ * 2);
  unsigned short* h_bf   = (unsigned short*)carve((size_t)B_ * H_ * 2);
  float* hbuf = (float*)carve((size_t)B_ * H_ * 4);
  float* cbuf = (float*)carve((size_t)B_ * H_ * 4);
  float* gbuf = (float*)carve((size_t)B_ * G4H * 4);

  // 1) weight conversion / repack (one-time; L2-resident afterwards)
  k_cvt_bf16<<<(NNM * C_) / 256, 256, 0, stream>>>(convw, cw_bf, NNM * C_);
  k_cvt_bf16<<<(V_ * E_) / 256, 256, 0, stream>>>(emb, emb_bf, V_ * E_);
  {
    long total = (long)256 * 64 * 32 * 16;        // ntiles*ksteps*32*16
    k_pack_B<<<(unsigned)((total + 255) / 256), 256, 0, stream>>>(Wg, wg_pack, G4H, 64, total);
  }
  {
    long total = (long)625 * 32 * 32 * 16;
    k_pack_B<<<(unsigned)((total + 255) / 256), 256, 0, stream>>>(Wo, wo_pack, V_, 32, total);
  }

  // 2) conv 1x1 GEMM + BN stats + pcf
  k_conv_gemm<<<(2048 * 8) / 8, 256, 0, stream>>>(cnn, cw_bf, y_bf);
  k_bn_stats<<<NNM, 256, 0, stream>>>(y_bf, chsum, chsq);
  k_bn_scale<<<2, 256, 0, stream>>>(chsum, chsq, gamma, beta, scale, shift);
  k_pcf<<<B_ * L_, 256, 0, stream>>>(y_bf, scale, shift, pcf_bf);

  // 3) LSTM time loop (stream-ordered; weights stay hot in L2)
  k_init_state<<<(B_ * H_) / 256, 256, 0, stream>>>(h_bf, cbuf);
  for (int t = 0; t < T_; ++t) {
    k_gates_gemm<<<32, 256, 0, stream>>>(emb_bf, pcf_bf, h_bf, wg_pack, toks, t, gbuf);
    k_lstm_cell<<<(B_ * H_) / 256, 256, 0, stream>>>(gbuf, bg, cbuf, hbuf, h_bf);
    k_logits_gemm<<<79, 256, 0, stream>>>(h_bf, wo_pack, bo, out, t);
  }
  k_copy_h<<<(B_ * H_) / 256, 256, 0, stream>>>(hbuf, out);
  (void)in_sizes; (void)n_in; (void)out_size; (void)ws_size;
}
